// ParallelComplexReciprocatorMixer_89060441850386
// MI455X (gfx1250) — matrix operand order repaired and checked by reading, not compiled, not run on hardware
//
#include <hip/hip_runtime.h>
#include <hip/hip_bf16.h>
#include <math.h>

// Problem constants
#define B_ 2
#define S_ 4096
#define H_ 768
#define E_ 2
#define R_ 4
#define D_ 256
#define FEAT_ 14
#define M_ (B_*S_)        // 8192 tokens
#define ERD_ (E_*R_*D_)   // 2048
#define CHN_ (B_*ERD_)    // 4096 scan channels
#define CHUNK_ 64
#define NCH_ (S_/CHUNK_)  // 64 chunks

typedef __attribute__((ext_vector_type(16))) __bf16 v16bf;
typedef __attribute__((ext_vector_type(8)))  float  v8f;

__device__ __forceinline__ unsigned short f2bf(float f) {
  unsigned int u = __float_as_uint(f);
  u += 0x7fffu + ((u >> 16) & 1u);   // round-to-nearest-even
  return (unsigned short)(u >> 16);
}

__device__ __forceinline__ v8f vzero8() {
  v8f z;
  for (int i = 0; i < 8; ++i) z[i] = 0.f;
  return z;
}

// ---- WMMA fragment loaders (CDNA5 16x16x32 bf16 layouts, ISA 7.12.2) ----
// A (16x32, row-major MxK): lane = M (lane&15); elems 0..7 = K(ko..ko+7),
// elems 8..15 = K(16+ko..), ko = 8*(lane>>4).
__device__ __forceinline__ v16bf load_a_frag(const unsigned short* __restrict__ A,
                                             int lda, int m0, int k0) {
  int lane = threadIdx.x & 31;
  int row  = m0 + (lane & 15);
  int ko   = (lane >> 4) * 8;
  const uint4* p0 = (const uint4*)(A + (size_t)row * lda + k0 + ko);
  const uint4* p1 = (const uint4*)(A + (size_t)row * lda + k0 + 16 + ko);
  union { v16bf v; uint4 q[2]; } u;
  u.q[0] = *p0;
  u.q[1] = *p1;
  return u.v;
}
// B (32x16, row-major KxN): lane = N (lane&15); elems 0..7 = K(ko..ko+7),
// elems 8..15 = K(16+ko..), ko = 8*(lane>>4).
__device__ __forceinline__ v16bf load_b_frag(const unsigned short* __restrict__ Bm,
                                             int ldb, int k0, int n0) {
  int lane = threadIdx.x & 31;
  int n  = n0 + (lane & 15);
  int ko = (lane >> 4) * 8;
  union { v16bf v; unsigned short s[16]; } u;
  for (int s = 0; s < 16; ++s) {
    int k = k0 + (s < 8 ? ko + s : 16 + ko + (s - 8));
    u.s[s] = Bm[(size_t)k * ldb + n];
  }
  return u.v;
}

// =======================================================================
// Prep kernels
// =======================================================================
__global__ void k_f2bf(const float* __restrict__ src, unsigned short* __restrict__ dst, int n) {
  int i = blockIdx.x * 256 + threadIdx.x;
  if (i < n) dst[i] = f2bf(src[i]);
}

// a = exp(-softplus(log_decay)) * e^{i theta}; also keep log|a| for a^p closed form
__global__ void k_decay(const float* __restrict__ log_decay, const float* __restrict__ theta,
                        float* __restrict__ lr, float* __restrict__ li, float* __restrict__ lm) {
  int i = blockIdx.x * 256 + threadIdx.x;
  if (i >= ERD_) return;
  float x  = log_decay[i];
  float sp = (x > 20.f) ? x : log1pf(__expf(x));
  float lmag = -sp;
  float mg = __expf(lmag);
  lr[i] = mg * __cosf(theta[i]);
  li[i] = mg * __sinf(theta[i]);
  lm[i] = lmag;
}

// Combined weights: Cr = Wsh_r*Wfus_r - Wsh_i*Wfus_i ; Ci = Wsh_r*Wfus_i + Wsh_i*Wfus_r
// shape (E*FEAT=28, H=768)
__global__ void k_combine(const float* __restrict__ Wsh_r, const float* __restrict__ Wsh_i,
                          const float* __restrict__ Wfus_r, const float* __restrict__ Wfus_i,
                          float* __restrict__ Cr, float* __restrict__ Ci) {
  int idx = blockIdx.x * 256 + threadIdx.x;
  if (idx >= 28 * H_) return;
  int ho = idx % H_;
  int ef = idx / H_;
  int e  = ef / FEAT_;
  float cr = 0.f, ci = 0.f;
  for (int h = 0; h < H_; ++h) {
    float wr = Wsh_r[(size_t)ef * H_ + h];
    float wi = Wsh_i[(size_t)ef * H_ + h];
    float fr = Wfus_r[(size_t)(e * H_ + h) * H_ + ho];
    float fi = Wfus_i[(size_t)(e * H_ + h) * H_ + ho];
    cr += wr * fr - wi * fi;
    ci += wr * fi + wi * fr;
  }
  Cr[idx] = cr;
  Ci[idx] = ci;
}

// Combined bias: bF = bfus + bsh (complex) pushed through Wfus
__global__ void k_biasfus(const float* __restrict__ bsh_r, const float* __restrict__ bsh_i,
                          const float* __restrict__ Wfus_r, const float* __restrict__ Wfus_i,
                          const float* __restrict__ bfus_r, const float* __restrict__ bfus_i,
                          float* __restrict__ bFr, float* __restrict__ bFi) {
  int ho = blockIdx.x * 256 + threadIdx.x;
  if (ho >= H_) return;
  float br = bfus_r[ho], bi = bfus_i[ho];
  for (int eh = 0; eh < E_ * H_; ++eh) {
    float srv = bsh_r[eh], siv = bsh_i[eh];
    float fr = Wfus_r[(size_t)eh * H_ + ho];
    float fi = Wfus_i[(size_t)eh * H_ + ho];
    br += srv * fr - siv * fi;
    bi += srv * fi + siv * fr;
  }
  bFr[ho] = br;
  bFi[ho] = bi;
}

// =======================================================================
// GEMM 1: sr/si = x @ Wsig_{r,i} + bsig  (M=8192, N=256 x2, K=768) bf16 WMMA
// wave tile: 16(M) x 64(N) for both r and i outputs (8 WMMAs / k-step)
// Direct fragment loads (operands are L2-resident; 192 MB L2).
// =======================================================================
__global__ void __launch_bounds__(256) k_gemm_sig(
    const unsigned short* __restrict__ xh,
    const unsigned short* __restrict__ Wr16, const unsigned short* __restrict__ Wi16,
    const float* __restrict__ bsr, const float* __restrict__ bsi,
    float* __restrict__ sr, float* __restrict__ si) {
  int wave = (int)(blockIdx.x * blockDim.x + threadIdx.x) >> 5;
  int lane = threadIdx.x & 31;
  int mt = wave >> 2;       // 0..511
  int ng = wave & 3;        // 0..3
  int m0 = mt * 16, n0 = ng * 64;
  v8f accR[4], accI[4];
  for (int j = 0; j < 4; ++j) { accR[j] = vzero8(); accI[j] = vzero8(); }

  for (int k0 = 0; k0 < H_; k0 += 32) {
    __builtin_prefetch(xh + (size_t)(m0 + (lane & 15)) * H_ + k0 + 64, 0, 1);
    v16bf a = load_a_frag(xh, H_, m0, k0);
#pragma unroll
    for (int j = 0; j < 4; ++j) {
      v16bf br = load_b_frag(Wr16, D_, k0, n0 + 16 * j);
      accR[j] = __builtin_amdgcn_wmma_f32_16x16x32_bf16(false, a, false, br,
                                                        (short)0, accR[j], false, false);
      v16bf bi = load_b_frag(Wi16, D_, k0, n0 + 16 * j);
      accI[j] = __builtin_amdgcn_wmma_f32_16x16x32_bf16(false, a, false, bi,
                                                        (short)0, accI[j], false, false);
    }
  }
  // C layout: VGPR v -> M = v + 8*(lane>=16), N = lane&15
  int col = lane & 15;
  int rbase = (lane >> 4) * 8;
#pragma unroll
  for (int j = 0; j < 4; ++j) {
    int n = n0 + 16 * j + col;
    float br_ = bsr[n], bi_ = bsi[n];
#pragma unroll
    for (int v = 0; v < 8; ++v) {
      int row = m0 + rbase + v;
      sr[(size_t)row * D_ + n] = accR[j][v] + br_;
      si[(size_t)row * D_ + n] = accI[j][v] + bi_;
    }
  }
}

// =======================================================================
// Scan phase A: per (channel, chunk) local recurrence h = a*h + u, h_in = 0
// =======================================================================
__global__ void k_scan_local(const float* __restrict__ sr, const float* __restrict__ si,
                             const float* __restrict__ lr, const float* __restrict__ li,
                             const float* __restrict__ binr, const float* __restrict__ bini,
                             float* __restrict__ hlr, float* __restrict__ hli,
                             float* __restrict__ carR, float* __restrict__ carI) {
  int idx = blockIdx.x * 256 + threadIdx.x;     // CHN_*NCH_ = 262144
  int d = idx & 255;
  int r = (idx >> 8) & 3;
  int e = (idx >> 10) & 1;
  int b = (idx >> 11) & 1;
  int chunk = idx >> 12;
  int erd = ((e * R_ + r) << 8) + d;
  float ar = lr[erd], ai = li[erd];
  float br = binr[erd], bi = bini[erd];
  float hr = 0.f, hi = 0.f;
  int s0 = chunk * CHUNK_;
  for (int j = 0; j < CHUNK_; ++j) {
    int m = b * S_ + s0 + j;
    float svr = sr[(size_t)m * D_ + d];
    float svi = si[(size_t)m * D_ + d];
    float ur = br * svr - bi * svi;
    float ui = br * svi + bi * svr;
    float nr = ar * hr - ai * hi + ur;
    float ni = ar * hi + ai * hr + ui;
    hr = nr; hi = ni;
    size_t ho = ((size_t)m * E_ + e) * (R_ * D_) + r * D_ + d;
    hlr[ho] = hr;
    hli[ho] = hi;
  }
  int chlin = idx & (CHN_ - 1);
  carR[chunk * CHN_ + chlin] = hr;
  carI[chunk * CHN_ + chlin] = hi;
}

// Scan phase B: per channel, scan chunk carries: p_{c+1} = a^64 * p_c + carry_c
__global__ void k_scan_carry(const float* __restrict__ lr, const float* __restrict__ li,
                             const float* __restrict__ carR, const float* __restrict__ carI,
                             float* __restrict__ preR, float* __restrict__ preI) {
  int chlin = blockIdx.x * 256 + threadIdx.x;   // 4096
  if (chlin >= CHN_) return;
  int erd = chlin & (ERD_ - 1);
  float ar = lr[erd], ai = li[erd];
  float pr64 = ar, pi64 = ai;
  for (int q = 0; q < 6; ++q) {                 // a^64 via 6 squarings
    float nr = pr64 * pr64 - pi64 * pi64;
    float ni = 2.f * pr64 * pi64;
    pr64 = nr; pi64 = ni;
  }
  float pr = 0.f, pi = 0.f;
  for (int c = 0; c < NCH_; ++c) {
    preR[c * CHN_ + chlin] = pr;
    preI[c * CHN_ + chlin] = pi;
    float cr = carR[c * CHN_ + chlin];
    float ci = carI[c * CHN_ + chlin];
    float nr = pr64 * pr - pi64 * pi + cr;
    float ni = pr64 * pi + pi64 * pr + ci;
    pr = nr; pi = ni;
  }
}

// =======================================================================
// Features: per (b,s,e) block: h = a^{j+1}*prefix + hloc, normalize over d,
// impression twist, reduce to 14 complex features.
// =======================================================================
__global__ void __launch_bounds__(256) k_features(
    const float* __restrict__ hlr, const float* __restrict__ hli,
    const float* __restrict__ preR, const float* __restrict__ preI,
    const float* __restrict__ lmag, const float* __restrict__ theta,
    float* __restrict__ Fr, float* __restrict__ Fi) {
  int blk = blockIdx.x;               // B*S*E = 16384
  int e = blk & 1;
  int s = (blk >> 1) & (S_ - 1);
  int b = blk >> 13;
  int m = b * S_ + s;
  int t = threadIdx.x;
  int r = t >> 6;
  int chunk = s >> 6;
  float pexp = (float)((s & 63) + 1);

  float hr[4], hi[4];
  float psum = 0.f;
#pragma unroll
  for (int k = 0; k < 4; ++k) {
    int d = (t & 63) + 64 * k;
    int erd = ((e * R_ + r) << 8) + d;
    int chlin = (((b * E_ + e) * R_ + r) << 8) + d;
    size_t ho = ((size_t)m * E_ + e) * (R_ * D_) + r * D_ + d;
    float mg = __expf(pexp * lmag[erd]);
    float th = pexp * theta[erd];
    float apr = mg * __cosf(th);
    float api = mg * __sinf(th);
    float pr = preR[chunk * CHN_ + chlin];
    float pi = preI[chunk * CHN_ + chlin];
    float xr = apr * pr - api * pi + hlr[ho];
    float xi = apr * pi + api * pr + hli[ho];
    hr[k] = xr; hi[k] = xi;
    psum += xr * xr + xi * xi;
  }
  __shared__ float red[256];
  __shared__ float invs[4];
  __shared__ float sums[4][7];
  red[t] = psum;
  __syncthreads();
  for (int off = 32; off >= 1; off >>= 1) {
    if ((t & 63) < off) red[t] += red[t + off];
    __syncthreads();
  }
  if ((t & 63) == 0) invs[r] = rsqrtf(red[t] / 256.f + 1e-6f);
  __syncthreads();
  float inv = invs[r];

  float pA = 0, pB = 0, pQr = 0, pQi = 0, pX = 0, pTr = 0, pTi = 0;
#pragma unroll
  for (int k = 0; k < 4; ++k) {
    float xr = hr[k] * inv;
    float xi = hi[k] * inv;
    float nr = xr * (1.f + 0.05f * tanhf(xi));
    float ni = xi * (1.f + 0.05f * tanhf(xr));
    pA += nr; pB += ni;
    pQr += nr * nr; pQi += ni * ni; pX += nr * ni;
    pTr += tanhf(nr); pTi += tanhf(ni);
  }
  float part[7] = {pA, pB, pQr, pQi, pX, pTr, pTi};
  for (int q = 0; q < 7; ++q) {
    __syncthreads();
    red[t] = part[q];
    __syncthreads();
    for (int off = 32; off >= 1; off >>= 1) {
      if ((t & 63) < off) red[t] += red[t + off];
      __syncthreads();
    }
    if ((t & 63) == 0) sums[r][q] = red[t];
  }
  __syncthreads();
  if (t == 0) {
    const float iD = 1.f / 256.f, iRD = 1.f / 1024.f;
    float gA = 0, gB = 0, gQr = 0, gQi = 0, gX = 0, gTr = 0, gTi = 0;
    for (int rr = 0; rr < 4; ++rr) {
      gA += sums[rr][0]; gB += sums[rr][1];
      gQr += sums[rr][2]; gQi += sums[rr][3];
      gX += sums[rr][4]; gTr += sums[rr][5]; gTi += sums[rr][6];
    }
    float* fr = Fr + (size_t)m * (E_ * FEAT_) + e * FEAT_;
    float* fi = Fi + (size_t)m * (E_ * FEAT_) + e * FEAT_;
    for (int rr = 0; rr < 4; ++rr) {
      fr[rr]     = sums[rr][0] * iD;                 // f1r
      fi[rr]     = sums[rr][1] * iD;                 // f1i
      fr[4 + rr] = (sums[rr][2] - sums[rr][3]) * iD; // f2r
      fi[4 + rr] = 2.f * sums[rr][4] * iD;           // f2i
    }
    fr[8]  = gA * iRD;          fi[8]  = gB * iRD;           // g1
    fr[9]  = (gQr - gQi) * iRD; fi[9]  = 2.f * gX * iRD;     // g2
    fr[10] = (gQr + gQi) * iRD; fi[10] = 0.f;                // g3
    fr[11] = gTr * iRD;         fi[11] = gTi * iRD;          // g4
    fr[12] = sqrtf(gQr * iRD + 1e-6f);
    fi[12] = sqrtf(gQi * iRD + 1e-6f);                       // g5
    fr[13] = gX * iRD;          fi[13] = 0.f;                // g6
  }
}

// =======================================================================
// Fuse: fur/fui = F @ C + bias (K=28 skinny), plus bf16 copies + fur*silu(fui)
// =======================================================================
__global__ void __launch_bounds__(256) k_fuse(
    const float* __restrict__ Fr, const float* __restrict__ Fi,
    const float* __restrict__ Cr, const float* __restrict__ Ci,
    const float* __restrict__ bFr, const float* __restrict__ bFi,
    float* __restrict__ fur, float* __restrict__ fui,
    unsigned short* __restrict__ fur16, unsigned short* __restrict__ fui16,
    unsigned short* __restrict__ furs16) {
  int blk = blockIdx.x;             // M_*3
  int m = blk / 3;
  int ho = (blk % 3) * 256 + threadIdx.x;
  __shared__ float sFr[28], sFi[28];
  if (threadIdx.x < 28) {
    sFr[threadIdx.x] = Fr[(size_t)m * 28 + threadIdx.x];
    sFi[threadIdx.x] = Fi[(size_t)m * 28 + threadIdx.x];
  }
  __syncthreads();
  float ar = bFr[ho], ai = bFi[ho];
#pragma unroll
  for (int k = 0; k < 28; ++k) {
    float cr = Cr[k * H_ + ho];
    float ci = Ci[k * H_ + ho];
    ar += sFr[k] * cr - sFi[k] * ci;
    ai += sFr[k] * ci + sFi[k] * cr;
  }
  size_t o = (size_t)m * H_ + ho;
  fur[o] = ar;
  fui[o] = ai;
  float sg = 1.f / (1.f + __expf(-ai));
  fur16[o] = f2bf(ar);
  fui16[o] = f2bf(ai);
  furs16[o] = f2bf(ar * (ai * sg));   // fur * silu(fui)
}

// =======================================================================
// GEMM 2 + epilogue: gate = sigmoid([x|fur|fui|fur*silu(fui)] @ Wgate + bgate)
// out = x + gate * fur.  M=8192, N=768, K=3072, bf16 WMMA.
// Workgroup tile 128(M)x64(N); the shared 32x64 B-tile is staged into LDS by
// GLOBAL_LOAD_ASYNC_TO_LDS_B128 (ASYNCcnt, double-buffered), ISA ch.10 / 08_async.
// =======================================================================
__global__ void __launch_bounds__(256) k_gemm_gate(
    const unsigned short* __restrict__ xh,
    const unsigned short* __restrict__ fur16, const unsigned short* __restrict__ fui16,
    const unsigned short* __restrict__ furs16,
    const unsigned short* __restrict__ Wg16, const float* __restrict__ bgate,
    const float* __restrict__ x, const float* __restrict__ fur,
    float* __restrict__ out) {
  __shared__ __align__(16) unsigned short bstage[2][32 * 64];   // 2 x 4KB
  int t = threadIdx.x;
  int w = t >> 5;
  int lane = t & 31;
  int bm = (int)blockIdx.x / 12;   // 0..63
  int bn = (int)blockIdx.x % 12;   // 0..11
  int m0 = bm * 128 + w * 16;
  int n0 = bn * 64;
  v8f acc[4];
  for (int j = 0; j < 4; ++j) acc[j] = vzero8();
  const unsigned short* srcs[4] = {xh, fur16, fui16, furs16};

  // Each thread asynchronously moves 16B of the B tile: row t>>3, 8 cols.
  int brow = t >> 3;               // 0..31 (K within tile)
  int bcol = (t & 7) * 8;          // 0..56
  auto issueB = [&](int k0, int buf) {
    unsigned goff = (unsigned)(((k0 + brow) * H_ + n0 + bcol) * 2);
    unsigned loff = (unsigned)(size_t)(void*)&bstage[buf][0] + (unsigned)(t * 16);
    asm volatile("global_load_async_to_lds_b128 %0, %1, %2"
                 :: "v"(loff), "v"(goff), "s"(Wg16) : "memory");
  };

  const int NSTEP = (4 * H_) / 32;   // 96
  issueB(0, 0);
#pragma unroll 1
  for (int ks = 0; ks < NSTEP; ++ks) {
    int buf = ks & 1;
    if (ks + 1 < NSTEP) {
      issueB((ks + 1) * 32, buf ^ 1);
      asm volatile("s_wait_asynccnt 0x1" ::: "memory");  // retire current tile
    } else {
      asm volatile("s_wait_asynccnt 0x0" ::: "memory");
    }
    __syncthreads();                 // current tile visible to all waves

    int blkA = ks / 24;              // which K-block source (x|fur|fui|furs)
    int kin = (ks % 24) * 32;
    __builtin_prefetch(srcs[blkA] + (size_t)(m0 + (lane & 15)) * H_ + kin + 64, 0, 1);
    v16bf a = load_a_frag(srcs[blkA], H_, m0, kin);
#pragma unroll
    for (int j = 0; j < 4; ++j) {
      // B frag from LDS: lane = N col 16*j+(lane&15); elems = K in A's order
      union { v16bf v; unsigned short s[16]; } ub;
      for (int s = 0; s < 16; ++s) {
        int kk = (s < 8 ? ((lane >> 4) * 8) + s : 16 + ((lane >> 4) * 8) + (s - 8));
        ub.s[s] = bstage[buf][kk * 64 + 16 * j + (lane & 15)];
      }
      acc[j] = __builtin_amdgcn_wmma_f32_16x16x32_bf16(false, a, false, ub.v,
                                                       (short)0, acc[j], false, false);
    }
    __syncthreads();                 // done reading before tile is overwritten
  }

  int col = lane & 15;
  int rbase = (lane >> 4) * 8;
#pragma unroll
  for (int j = 0; j < 4; ++j) {
    int n = n0 + 16 * j + col;
    float bg = bgate[n];
#pragma unroll
    for (int v = 0; v < 8; ++v) {
      int row = m0 + rbase + v;
      float e = acc[j][v] + bg;
      float g = 1.f / (1.f + __expf(-e));
      size_t o = (size_t)row * H_ + n;
      out[o] = x[o] + g * fur[o];
    }
  }
}

// =======================================================================
extern "C" void kernel_launch(void* const* d_in, const int* in_sizes, int n_in,
                              void* d_out, int out_size, void* d_ws, size_t ws_size,
                              hipStream_t stream) {
  const float* x       = (const float*)d_in[0];
  const float* Wsig_r  = (const float*)d_in[1];
  const float* Wsig_i  = (const float*)d_in[2];
  const float* bsig_r  = (const float*)d_in[3];
  const float* bsig_i  = (const float*)d_in[4];
  const float* log_dec = (const float*)d_in[5];
  const float* theta   = (const float*)d_in[6];
  const float* Bin_r   = (const float*)d_in[7];
  const float* Bin_i   = (const float*)d_in[8];
  const float* Wsh_r   = (const float*)d_in[9];
  const float* Wsh_i   = (const float*)d_in[10];
  const float* bsh_r   = (const float*)d_in[11];
  const float* bsh_i   = (const float*)d_in[12];
  const float* Wfus_r  = (const float*)d_in[13];
  const float* Wfus_i  = (const float*)d_in[14];
  const float* bfus_r  = (const float*)d_in[15];
  const float* bfus_i  = (const float*)d_in[16];
  const float* Wgate   = (const float*)d_in[17];
  const float* bgate   = (const float*)d_in[18];
  float* out = (float*)d_out;

  char* w = (char*)d_ws;
  size_t off = 0;
  auto alloc = [&](size_t bytes) -> void* {
    off = (off + 255) & ~(size_t)255;
    void* p = w + off;
    off += bytes;
    return p;
  };

  unsigned short* xh    = (unsigned short*)alloc((size_t)M_ * H_ * 2);
  unsigned short* Wsr16 = (unsigned short*)alloc((size_t)H_ * D_ * 2);
  unsigned short* Wsi16 = (unsigned short*)alloc((size_t)H_ * D_ * 2);
  unsigned short* Wg16  = (unsigned short*)alloc((size_t)4 * H_ * H_ * 2);
  float* lr   = (float*)alloc(ERD_ * 4);
  float* li   = (float*)alloc(ERD_ * 4);
  float* lm   = (float*)alloc(ERD_ * 4);
  float* sr   = (float*)alloc((size_t)M_ * D_ * 4);
  float* si   = (float*)alloc((size_t)M_ * D_ * 4);
  float* Cr   = (float*)alloc(28 * H_ * 4);
  float* Ci   = (float*)alloc(28 * H_ * 4);
  float* bFr  = (float*)alloc(H_ * 4);
  float* bFi  = (float*)alloc(H_ * 4);
  float* Fr   = (float*)alloc((size_t)M_ * 28 * 4);
  float* Fi   = (float*)alloc((size_t)M_ * 28 * 4);
  float* carR = (float*)alloc((size_t)NCH_ * CHN_ * 4);
  float* carI = (float*)alloc((size_t)NCH_ * CHN_ * 4);
  float* preR = (float*)alloc((size_t)NCH_ * CHN_ * 4);
  float* preI = (float*)alloc((size_t)NCH_ * CHN_ * 4);
  float* hlR  = (float*)alloc((size_t)M_ * ERD_ * 4);   // 67 MB
  float* hlI  = (float*)alloc((size_t)M_ * ERD_ * 4);   // 67 MB
  // fur/fui & bf16 copies alias the hl region (only live after k_features).
  char* ali = (char*)hlR;
  float* fur = (float*)ali;              ali += (size_t)M_ * H_ * 4;
  float* fui = (float*)ali;              ali += (size_t)M_ * H_ * 4;
  unsigned short* fur16  = (unsigned short*)ali; ali += (size_t)M_ * H_ * 2;
  unsigned short* fui16  = (unsigned short*)ali; ali += (size_t)M_ * H_ * 2;
  unsigned short* furs16 = (unsigned short*)ali;

  // ---- prep ----
  k_f2bf<<<(M_ * H_ + 255) / 256, 256, 0, stream>>>(x, xh, M_ * H_);
  k_f2bf<<<(H_ * D_ + 255) / 256, 256, 0, stream>>>(Wsig_r, Wsr16, H_ * D_);
  k_f2bf<<<(H_ * D_ + 255) / 256, 256, 0, stream>>>(Wsig_i, Wsi16, H_ * D_);
  k_f2bf<<<(4 * H_ * H_ + 255) / 256, 256, 0, stream>>>(Wgate, Wg16, 4 * H_ * H_);
  k_decay<<<(ERD_ + 255) / 256, 256, 0, stream>>>(log_dec, theta, lr, li, lm);
  k_combine<<<(28 * H_ + 255) / 256, 256, 0, stream>>>(Wsh_r, Wsh_i, Wfus_r, Wfus_i, Cr, Ci);
  k_biasfus<<<(H_ + 255) / 256, 256, 0, stream>>>(bsh_r, bsh_i, Wfus_r, Wfus_i,
                                                  bfus_r, bfus_i, bFr, bFi);
  // ---- signal GEMM (WMMA bf16) ----
  k_gemm_sig<<<256, 256, 0, stream>>>(xh, Wsr16, Wsi16, bsig_r, bsig_i, sr, si);
  // ---- chunked linear recurrence ----
  k_scan_local<<<(CHN_ * NCH_) / 256, 256, 0, stream>>>(sr, si, lr, li, Bin_r, Bin_i,
                                                        hlR, hlI, carR, carI);
  k_scan_carry<<<CHN_ / 256, 256, 0, stream>>>(lr, li, carR, carI, preR, preI);
  // ---- features (fused scan fix-up + norm + impression + reductions) ----
  k_features<<<B_ * S_ * E_, 256, 0, stream>>>(hlR, hlI, preR, preI, lm, theta, Fr, Fi);
  // ---- fused Wsh∘Wfus application (K=28) ----
  k_fuse<<<M_ * 3, 256, 0, stream>>>(Fr, Fi, Cr, Ci, bFr, bFi,
                                     fur, fui, fur16, fui16, furs16);
  // ---- gate GEMM (WMMA bf16, async LDS staging) + epilogue ----
  k_gemm_gate<<<768, 256, 0, stream>>>(xh, fur16, fui16, furs16, Wg16, bgate,
                                       x, fur, out);
  (void)in_sizes; (void)n_in; (void)out_size; (void)ws_size;
}
